// GCNEncoderNetwork_74071005987301
// MI455X (gfx1250) — matrix operand order repaired and checked by reading, not compiled, not run on hardware
//
#include <hip/hip_runtime.h>
#include <hip/hip_bf16.h>

typedef _Float16 f16;
typedef __attribute__((ext_vector_type(16))) _Float16 v16h;
typedef __attribute__((ext_vector_type(8)))  _Float16 v8h;
typedef __attribute__((ext_vector_type(4)))  _Float16 v4h;
typedef __attribute__((ext_vector_type(8)))  float    v8f;

#define DH    128   // feature dim for every layer (D_IN = D_HID = 2*LATENT = 128)
#define PITCH 136   // LDS row pitch in halfs: 272 B = 68 dwords -> conflict-free, 16B-aligned
#define MT    8     // M-tiles (of 16 rows) processed per block

// truncate flat shared-aperture address to raw LDS byte offset (aperture lives in addr[63:32])
__device__ __forceinline__ unsigned lds_off_of(const void* p) {
  return (unsigned)(unsigned long long)p;
}

// one 16-byte async DMA transfer: LDS[lds_byte_off] = MEM[gaddr]
__device__ __forceinline__ void async_copy_b128(unsigned lds_byte_off, const void* gaddr) {
  asm volatile("global_load_async_to_lds_b128 %0, %1, off"
               :: "v"(lds_byte_off), "v"((unsigned long long)gaddr)
               : "memory");
}

__device__ __forceinline__ void wait_asynccnt0() {
  asm volatile("s_wait_asynccnt 0x0" ::: "memory");
}

// ---------------- utility kernels ----------------

__global__ void k_zero(float* __restrict__ p, int n) {
  int i = blockIdx.x * blockDim.x + threadIdx.x;
  if (i < n) p[i] = 0.0f;
}

__global__ void k_f32_to_f16(const float* __restrict__ in, f16* __restrict__ out, int n) {
  int i = blockIdx.x * blockDim.x + threadIdx.x;
  if (i < n) out[i] = (f16)in[i];
}

// Wt[n*128 + k] = (f16) W[k*128 + n]  -- transposed cast so B fragments are contiguous
__global__ void k_w_transpose_f16(const float* __restrict__ W, f16* __restrict__ Wt) {
  int i = blockIdx.x * blockDim.x + threadIdx.x;
  if (i >= DH * DH) return;
  const int n = i >> 7;
  const int k = i & 127;
  Wt[i] = (f16)W[k * DH + n];
}

// deg[v] = number of incoming edges (self loop added later as +1)
__global__ void k_deg(const long long* __restrict__ dst, float* __restrict__ deg, int nE) {
  int e = blockIdx.x * blockDim.x + threadIdx.x;
  if (e < nE) atomicAdd(&deg[dst[e]], 1.0f);
}

// dinv[v] = rsqrt(deg[v] + 1)
__global__ void k_dinv(float* __restrict__ deg_dinv, int n) {
  int i = blockIdx.x * blockDim.x + threadIdx.x;
  if (i < n) deg_dinv[i] = rsqrtf(deg_dinv[i] + 1.0f);
}

// ---------------- WMMA GEMM: G[N,128] = A[N,128] @ W[128,128] ----------------
// A row-major f16; Wt = W transposed (Wt[n,k] = W[k,n]) f16; f32 accumulate; f16 out.

__global__ __launch_bounds__(256)
void k_gemm128_wmma(const f16* __restrict__ A, const f16* __restrict__ Wt,
                    f16* __restrict__ G, int nrows) {
  __shared__ __align__(16) f16 sW[DH * PITCH];  // ~34.8 KB transposed weights
  __shared__ __align__(16) f16 sA[16 * PITCH];  // ~4.3 KB A tile

  const int tid  = threadIdx.x;
  const bool full = (int)(blockIdx.x + 1) * (MT * 16) <= nrows;  // uniform per block

  // stage transposed weights once per block via async DMA:
  // 128 rows * 16 b128-chunks = 2048 transfers, 8 per thread
  for (int i = tid; i < DH * (DH / 8); i += 256) {
    const int n = i >> 4;          // Wt row (output column)
    const int c = (i & 15) << 3;   // half offset within row
    async_copy_b128(lds_off_of(sW + n * PITCH + c), Wt + n * DH + c);
  }

  const int lane = tid & 31;
  const int wave = tid >> 5;    // 0..7 -> 16-column output tile
  const int hi   = lane >> 4;
  const int lo   = lane & 15;
  const int n0   = wave * 16;

  for (int t = 0; t < MT; ++t) {
    const int m0 = (blockIdx.x * MT + t) * 16;
    if (m0 >= nrows) break;                       // uniform per block

    __syncthreads();  // previous tile's reads done before sA overwrite

    if (full) {
      // async stage A tile: 16 rows * 16 b128-chunks = 256 transfers, 1 per thread
      const int r = tid >> 4;
      const int c = (tid & 15) << 3;
      async_copy_b128(lds_off_of(sA + r * PITCH + c), A + (size_t)(m0 + r) * DH + c);
    } else {
      // boundary block: guarded synchronous staging with zero fill
      const int rowsLeft = nrows - m0;
      for (int i = tid; i < 16 * (DH / 4); i += 256) {
        const int r = i >> 5;
        const int c = (i & 31) << 2;
        uint2 z; z.x = 0u; z.y = 0u;
        *(uint2*)(sA + r * PITCH + c) =
            (r < rowsLeft) ? *(const uint2*)(A + (size_t)(m0 + r) * DH + c) : z;
      }
    }
    wait_asynccnt0();   // covers sA tile (and sW on first iteration)
    __syncthreads();

    v8f acc = {};
    #pragma unroll
    for (int kc = 0; kc < 4; ++kc) {              // K = 128 in chunks of 32
      const int k0 = kc * 32;
      // A 16x32 fragment: lane row M=lo; halves hold K 8*hi..+7 then 16+8*hi..+7
      const v8h a0 = *(const v8h*)(sA + lo * PITCH + k0 + 8 * hi);
      const v8h a1 = *(const v8h*)(sA + lo * PITCH + k0 + 16 + 8 * hi);
      // B 32x16 fragment: lane col N=n0+lo; halves hold K 8*hi..+7 then 16+8*hi..+7
      const v8h b0 = *(const v8h*)(sW + (n0 + lo) * PITCH + k0 + 8 * hi);
      const v8h b1 = *(const v8h*)(sW + (n0 + lo) * PITCH + k0 + 16 + 8 * hi);
      const v16h a = __builtin_shufflevector(a0, a1, 0, 1, 2, 3, 4, 5, 6, 7,
                                             8, 9, 10, 11, 12, 13, 14, 15);
      const v16h b = __builtin_shufflevector(b0, b1, 0, 1, 2, 3, 4, 5, 6, 7,
                                             8, 9, 10, 11, 12, 13, 14, 15);
      acc = __builtin_amdgcn_wmma_f32_16x16x32_f16(
          /*neg_a=*/false, a, /*neg_b=*/false, b,
          /*c_mod=*/(short)0, acc, /*reuse_a=*/false, /*reuse_b=*/false);
    }

    // C/D layout: VGPR r holds row M = r + 8*hi, column N = lo
    f16* gout = G + (size_t)(m0 + 8 * hi) * DH + n0 + lo;
    if (full) {
      #pragma unroll
      for (int r = 0; r < 8; ++r) gout[(size_t)r * DH] = (f16)acc[r];
    } else {
      #pragma unroll
      for (int r = 0; r < 8; ++r) {
        if (m0 + r + 8 * hi < nrows) gout[(size_t)r * DH] = (f16)acc[r];
      }
    }
  }
}

// ---------------- edge scatter: AGG[dst] += G[src] * dinv[src]*dinv[dst] ----------------

__global__ __launch_bounds__(256)
void k_edge_scatter(const long long* __restrict__ src, const long long* __restrict__ dst,
                    const float* __restrict__ dinv, const f16* __restrict__ G,
                    float* __restrict__ AGG, int nE) {
  const int e = blockIdx.x * 8 + (threadIdx.x >> 5);   // one wave per edge
  if (e >= nE) return;
  const int lane = threadIdx.x & 31;                   // lane covers 4 features
  const long long s = src[e];
  const long long d = dst[e];
  const float w = dinv[s] * dinv[d];
  const v4h g = *(const v4h*)(G + (size_t)s * DH + lane * 4);  // coalesced 8B per lane
  float* o = AGG + (size_t)d * DH + lane * 4;
  atomicAdd(o + 0, (float)g[0] * w);
  atomicAdd(o + 1, (float)g[1] * w);
  atomicAdd(o + 2, (float)g[2] * w);
  atomicAdd(o + 3, (float)g[3] * w);
}

// ---------------- finalize: self-loop + bias (+ ReLU -> f16, or pooled readout) ----------------

__global__ void k_finalize_relu(const float* __restrict__ AGG, const f16* __restrict__ G,
                                const float* __restrict__ dinv, const float* __restrict__ bias,
                                f16* __restrict__ H, int n) {  // n = N*DH
  int i = blockIdx.x * blockDim.x + threadIdx.x;
  if (i >= n) return;
  const int v = i >> 7;
  const int f = i & 127;
  const float di = dinv[v];
  const float val = AGG[i] + (float)G[i] * di * di + bias[f];
  H[i] = (f16)fmaxf(val, 0.0f);
}

__global__ void k_finalize_pool(const float* __restrict__ AGG, const f16* __restrict__ G,
                                const float* __restrict__ dinv, const float* __restrict__ bias,
                                const long long* __restrict__ batch,
                                float* __restrict__ pool, float* __restrict__ cnt, int n) {
  int i = blockIdx.x * blockDim.x + threadIdx.x;
  if (i >= n) return;
  const int v = i >> 7;
  const int f = i & 127;
  const float di = dinv[v];
  const float val = AGG[i] + (float)G[i] * di * di + bias[f];
  const long long g = batch[v];
  atomicAdd(&pool[(size_t)g * DH + f], val);
  if (f == 0) atomicAdd(&cnt[g], 1.0f);
}

__global__ void k_pool_div(const float* __restrict__ pool, const float* __restrict__ cnt,
                           float* __restrict__ out, int n) {
  int i = blockIdx.x * blockDim.x + threadIdx.x;
  if (i < n) out[i] = pool[i] / fmaxf(cnt[i >> 7], 1.0f);
}

// ---------------- driver ----------------

extern "C" void kernel_launch(void* const* d_in, const int* in_sizes, int n_in,
                              void* d_out, int out_size, void* d_ws, size_t ws_size,
                              hipStream_t stream) {
  (void)n_in; (void)ws_size;
  const float*     x     = (const float*)d_in[0];
  const long long* ei    = (const long long*)d_in[1];  // int64 in reference
  const long long* batch = (const long long*)d_in[2];  // int64
  const float* W1 = (const float*)d_in[3];
  const float* b1 = (const float*)d_in[4];
  const float* W2 = (const float*)d_in[5];
  const float* b2 = (const float*)d_in[6];
  const float* W3 = (const float*)d_in[7];
  const float* b3 = (const float*)d_in[8];

  const int N  = in_sizes[0] / DH;   // 100000
  const int E  = in_sizes[1] / 2;    // 1.6M
  const int NG = out_size / DH;      // 512
  const long long* src = ei;         // edge_index[0]
  const long long* dst = ei + E;     // edge_index[1]

  // workspace carve-out (256B aligned)
  char* ws = (char*)d_ws;
  auto carve = [&](size_t bytes) -> char* {
    char* p = ws; ws += (bytes + 255) & ~(size_t)255; return p;
  };
  float* dinv = (float*)carve((size_t)N * 4);            // deg, then rsqrt in place
  f16*   H    = (f16*)  carve((size_t)N * DH * 2);       // GEMM input (f16)
  f16*   G    = (f16*)  carve((size_t)N * DH * 2);       // GEMM output (f16)
  float* AGG  = (float*)carve((size_t)N * DH * 4);       // scatter accumulator
  f16*   W1t  = (f16*)  carve((size_t)DH * DH * 2);      // transposed f16 weights
  f16*   W2t  = (f16*)  carve((size_t)DH * DH * 2);
  f16*   W3t  = (f16*)  carve((size_t)DH * DH * 2);
  float* pool = (float*)carve((size_t)NG * DH * 4);
  float* cnt  = (float*)carve((size_t)NG * 4);

  const int T = 256;
  auto cdiv = [](long long a, long long b) { return (int)((a + b - 1) / b); };
  const int NF = N * DH;  // 12.8M, fits in int

  // 1) degree -> dinv
  k_zero<<<cdiv(N, T), T, 0, stream>>>(dinv, N);
  k_deg <<<cdiv(E, T), T, 0, stream>>>(dst, dinv, E);
  k_dinv<<<cdiv(N, T), T, 0, stream>>>(dinv, N);

  // 2) transpose-cast weights, cast input features to f16
  k_w_transpose_f16<<<cdiv(DH * DH, T), T, 0, stream>>>(W1, W1t);
  k_w_transpose_f16<<<cdiv(DH * DH, T), T, 0, stream>>>(W2, W2t);
  k_w_transpose_f16<<<cdiv(DH * DH, T), T, 0, stream>>>(W3, W3t);
  k_f32_to_f16<<<cdiv(NF, T), T, 0, stream>>>(x, H, NF);

  const int gemmGrid = cdiv(N, 16 * MT);
  const f16*   Wt[3] = {W1t, W2t, W3t};
  const float* bb[3] = {b1, b2, b3};

  for (int layer = 0; layer < 3; ++layer) {
    k_gemm128_wmma<<<gemmGrid, 256, 0, stream>>>(H, Wt[layer], G, N);
    k_zero<<<cdiv(NF, T), T, 0, stream>>>(AGG, NF);
    k_edge_scatter<<<cdiv(E, 8), 256, 0, stream>>>(src, dst, dinv, G, AGG, E);
    if (layer < 2) {
      k_finalize_relu<<<cdiv(NF, T), T, 0, stream>>>(AGG, G, dinv, bb[layer], H, NF);
    } else {
      k_zero<<<cdiv(NG * DH, T), T, 0, stream>>>(pool, NG * DH);
      k_zero<<<cdiv(NG, T), T, 0, stream>>>(cnt, NG);
      k_finalize_pool<<<cdiv(NF, T), T, 0, stream>>>(AGG, G, dinv, b3, batch, pool, cnt, NF);
      k_pool_div<<<cdiv(NG * DH, T), T, 0, stream>>>(pool, cnt, (float*)d_out, NG * DH);
    }
  }
}
